// GPTLanguageModel_82875688943922
// MI455X (gfx1250) — matrix operand order repaired and checked by reading, not compiled, not run on hardware
//
#include <hip/hip_runtime.h>
#include <cmath>

// ---------------------------------------------------------------------------
// GPT forward for MI455X (gfx1250): bf16 WMMA GEMMs with f32 accumulate.
// ---------------------------------------------------------------------------

typedef __bf16 bf16;
typedef __attribute__((ext_vector_type(8)))  __bf16 v8bf;
typedef __attribute__((ext_vector_type(16))) __bf16 v16bf;
typedef __attribute__((ext_vector_type(8)))  float  v8f;

#define E_DIM     768
#define NHEAD     12
#define HDIM      64
#define NLAYER    6
#define BSZ       4
#define TSEQ      1024
#define NTOK      4096          /* BSZ*TSEQ */
#define VOCAB_N   50257
#define VOCAB_PAD 50264         /* next multiple of 8 for aligned 16B loads */
#define FF        3072

#define BM   128
#define BN   128
#define BKK  32
#define LPAD 8
#define LSTR (BKK + LPAD)       /* 40 bf16 = 80B row stride, 16B aligned */

// ------------------------------- helpers -----------------------------------

__global__ __launch_bounds__(256) void embed_kernel(
    const int* __restrict__ idx, const float* __restrict__ tok,
    const float* __restrict__ pos, float* __restrict__ x) {
  long i = (long)blockIdx.x * 256 + threadIdx.x;   // over NTOK*E_DIM
  int row = (int)(i / E_DIM);
  int e   = (int)(i % E_DIM);
  int t   = row % TSEQ;
  x[i] = tok[(long)idx[row] * E_DIM + e] + pos[(long)t * E_DIM + e];
}

__global__ __launch_bounds__(256) void cvt_kernel(
    const float* __restrict__ in, bf16* __restrict__ out, long n) {
  long i = (long)blockIdx.x * 256 + threadIdx.x;
  if (i < n) out[i] = (bf16)in[i];
}

__global__ __launch_bounds__(256) void cvt_pad_kernel(
    const float* __restrict__ in, bf16* __restrict__ out,
    int rows, int cols, int colsPad) {
  long i = (long)blockIdx.x * 256 + threadIdx.x;
  long total = (long)rows * colsPad;
  if (i >= total) return;
  int r = (int)(i / colsPad);
  int c = (int)(i % colsPad);
  out[i] = (c < cols) ? (bf16)in[(long)r * cols + c] : (bf16)0.0f;
}

// pack (L,H,E,HD) x {Wq,Wk,Wv} f32 -> (L, E, 3E) bf16, col = sel*768+h*64+d
__global__ __launch_bounds__(256) void pack_qkv_kernel(
    const float* __restrict__ Wq, const float* __restrict__ Wk,
    const float* __restrict__ Wv, bf16* __restrict__ out) {
  long i = (long)blockIdx.x * 256 + threadIdx.x;
  long per = (long)E_DIM * 3 * E_DIM;
  if (i >= (long)NLAYER * per) return;
  int  l   = (int)(i / per);
  long rem = i % per;
  int  e   = (int)(rem / (3 * E_DIM));
  int  n   = (int)(rem % (3 * E_DIM));
  int  sel = n / E_DIM;
  int  r   = n % E_DIM;
  int  h   = r / HDIM;
  int  d   = r % HDIM;
  const float* W = (sel == 0) ? Wq : (sel == 1) ? Wk : Wv;
  out[i] = (bf16)W[((((long)l * NHEAD + h) * E_DIM) + e) * HDIM + d];
}

__global__ __launch_bounds__(256) void layernorm_kernel(
    const float* __restrict__ x, const float* __restrict__ gamma,
    const float* __restrict__ beta, bf16* __restrict__ out) {
  int row = blockIdx.x;
  int tid = threadIdx.x;
  const float* xr = x + (long)row * E_DIM;
  float s = 0.f, s2 = 0.f;
  for (int c = tid; c < E_DIM; c += 256) { float v = xr[c]; s += v; s2 += v * v; }
  __shared__ float r1[256], r2[256];
  r1[tid] = s; r2[tid] = s2;
  __syncthreads();
  for (int off = 128; off > 0; off >>= 1) {
    if (tid < off) { r1[tid] += r1[tid + off]; r2[tid] += r2[tid + off]; }
    __syncthreads();
  }
  float mean = r1[0] * (1.0f / E_DIM);
  float var  = r2[0] * (1.0f / E_DIM) - mean * mean;
  float rstd = rsqrtf(var + 1e-5f);
  for (int c = tid; c < E_DIM; c += 256)
    out[(long)row * E_DIM + c] = (bf16)((xr[c] - mean) * rstd * gamma[c] + beta[c]);
}

// one 256-thread block per score row; causal mask + softmax -> bf16 probs
__global__ __launch_bounds__(256) void softmax_kernel(
    const float* __restrict__ S, bf16* __restrict__ P) {
  long row = blockIdx.x;                  // over BSZ*NHEAD*TSEQ
  int  t   = (int)(row % TSEQ);
  const float* s = S + row * (long)TSEQ;
  bf16*        p = P + row * (long)TSEQ;
  int tid = threadIdx.x;
  float vals[4];
  float mx = -3.4e38f;
#pragma unroll
  for (int i = 0; i < 4; ++i) {
    int c = tid + i * 256;
    vals[i] = (c <= t) ? s[c] : -3.4e38f;
    mx = fmaxf(mx, vals[i]);
  }
  __shared__ float red[256];
  red[tid] = mx; __syncthreads();
  for (int off = 128; off > 0; off >>= 1) {
    if (tid < off) red[tid] = fmaxf(red[tid], red[tid + off]);
    __syncthreads();
  }
  mx = red[0];
  __syncthreads();
  float sum = 0.f;
#pragma unroll
  for (int i = 0; i < 4; ++i) {
    int c = tid + i * 256;
    vals[i] = (c <= t) ? __expf(vals[i] - mx) : 0.f;
    sum += vals[i];
  }
  red[tid] = sum; __syncthreads();
  for (int off = 128; off > 0; off >>= 1) {
    if (tid < off) red[tid] += red[tid + off];
    __syncthreads();
  }
  float inv = 1.0f / red[0];
#pragma unroll
  for (int i = 0; i < 4; ++i) {
    int c = tid + i * 256;
    p[c] = (bf16)(vals[i] * inv);
  }
}

// ------------------------------ WMMA GEMM ----------------------------------
// C[z] = op(A[z]) * B[z] (+bias)(ReLU?)(+Res), A: MxK bf16 row-major,
// B: KxN bf16 row-major (btrans=0) or NxK row-major, i.e. A*B^T (btrans=1).
// blockIdx.z decomposed as (bz,hz)=(z/hdec, z%hdec) with per-operand strides,
// so strided sub-views (per batch b and head h) of packed tensors work.
// 256 threads = 8 wave32 waves; 128x128 tile; K-step 32; 8 WMMAs/wave/step.
__global__ __launch_bounds__(256) void wmma_gemm_bf16(
    const bf16* __restrict__ A, long lda, long sbA, long shA,
    const bf16* __restrict__ B, long ldb, long sbB, long shB, int btrans,
    float* __restrict__ Cf, bf16* __restrict__ Ch, long ldc, long sbC, long shC,
    const float* __restrict__ bias,
    const float* __restrict__ Res, long ldr, long sbR, long shR,
    int M, int N, int K, float scale, int relu, int hdec) {
  __shared__ __align__(16) bf16 As[BM][LSTR];
  __shared__ __align__(16) bf16 Bs[BN][LSTR];   // stored transposed: Bs[n][k]

  int tid  = threadIdx.x;
  int wave = tid >> 5;
  int lane = tid & 31;
  int m    = lane & 15;      // A row / B col / C col within 16x16 tile
  int g    = lane >> 4;      // lane group

  int bz = blockIdx.z / hdec;
  int hz = blockIdx.z % hdec;

  const bf16* Ab = A + bz * sbA + hz * shA;
  const bf16* Bb = B + bz * sbB + hz * shB;

  int rowBase = blockIdx.y * BM;
  int colBase = blockIdx.x * BN;

  v8f acc[8];
#pragma unroll
  for (int i = 0; i < 8; ++i)
#pragma unroll
    for (int j = 0; j < 8; ++j) acc[i][j] = 0.f;

  for (int k0 = 0; k0 < K; k0 += BKK) {
    // ---- stage A tile (BM x BKK), coalesced 16B loads ----
#pragma unroll
    for (int p = 0; p < 2; ++p) {
      int flat = (p * 256 + tid) * 8;          // over 128*32
      int r = flat >> 5;
      int c = flat & 31;
      const bf16* src = Ab + (long)(rowBase + r) * lda + (k0 + c);
      *(v8bf*)(&As[r][c]) = *(const v8bf*)src;
    }
    // ---- stage B tile into transposed LDS: Bs[n][k] ----
    if (btrans) {                              // B is NxK: direct row copy
#pragma unroll
      for (int p = 0; p < 2; ++p) {
        int flat = (p * 256 + tid) * 8;        // over 128*32
        int n = flat >> 5;
        int k = flat & 31;
        v8bf val;
        if (colBase + n < N) {
          val = *(const v8bf*)(Bb + (long)(colBase + n) * ldb + (k0 + k));
        } else {
#pragma unroll
          for (int j = 0; j < 8; ++j) val[j] = (bf16)0.0f;
        }
        *(v8bf*)(&Bs[n][k]) = val;
      }
    } else {                                   // B is KxN: transpose on store
#pragma unroll
      for (int p = 0; p < 2; ++p) {
        int flat = (p * 256 + tid) * 8;        // over 32*128
        int k = flat >> 7;
        int n = flat & 127;
        const bf16* src = Bb + (long)(k0 + k) * ldb + (colBase + n);
        bf16 tmp[8];
        if (colBase + n + 7 < N) {
          *(v8bf*)tmp = *(const v8bf*)src;
        } else {
#pragma unroll
          for (int j = 0; j < 8; ++j)
            tmp[j] = (colBase + n + j < N) ? src[j] : (bf16)0.0f;
        }
#pragma unroll
        for (int j = 0; j < 8; ++j) Bs[n + j][k] = tmp[j];
      }
    }
    __syncthreads();

    // hint next A tile toward the caches while we do math on this one
    if (k0 + BKK < K)
      __builtin_prefetch(Ab + (long)(rowBase + (tid >> 1)) * lda + (k0 + BKK), 0, 1);

    // ---- compute: wave owns rows [wave*16, wave*16+16), all 128 cols ----
    // A fragment (ISA 16-bit 16x32 layout): lane row = lane%16,
    // elems 0..7 -> K = g*8+0..7, elems 8..15 -> K = 16+g*8+0..7.
    v8bf alo = *(const v8bf*)(&As[wave * 16 + m][g * 8]);
    v8bf ahi = *(const v8bf*)(&As[wave * 16 + m][16 + g * 8]);
    v16bf a;
#pragma unroll
    for (int i = 0; i < 8; ++i) { a[i] = alo[i]; a[i + 8] = ahi[i]; }
#pragma unroll
    for (int nt = 0; nt < 8; ++nt) {
      // B fragment (32x16): lane col = lane%16, elems 0..7 -> K = g*8+0..7, elems 8..15 -> K = 16+g*8+0..7.
      v8bf blo = *(const v8bf*)(&Bs[nt * 16 + m][g * 8]);
      v8bf bhi = *(const v8bf*)(&Bs[nt * 16 + m][16 + g * 8]);
      v16bf b;
#pragma unroll
      for (int i = 0; i < 8; ++i) { b[i] = blo[i]; b[i + 8] = bhi[i]; }
      acc[nt] = __builtin_amdgcn_wmma_f32_16x16x32_bf16(
          false, a, false, b, (short)0, acc[nt], false, false);
    }
    __syncthreads();
  }

  // ---- epilogue: C 16x16 f32 layout: lane col = lane%16, VGPR r -> row r+8g
  float*       Cfp = Cf ? (Cf + bz * sbC + hz * shC) : (float*)0;
  bf16*        Chp = Ch ? (Ch + bz * sbC + hz * shC) : (bf16*)0;
  const float* Rp  = Res ? (Res + bz * sbR + hz * shR) : (const float*)0;

#pragma unroll
  for (int nt = 0; nt < 8; ++nt) {
    int col = colBase + nt * 16 + m;
    if (col >= N) continue;
    float bv = bias ? bias[col] : 0.f;
#pragma unroll
    for (int r = 0; r < 8; ++r) {
      int row = rowBase + wave * 16 + r + 8 * g;
      float v = acc[nt][r] * scale + bv;
      if (relu) v = v > 0.f ? v : 0.f;
      if (Rp)  v += Rp[(long)row * ldr + col];
      if (Cfp) Cfp[(long)row * ldc + col] = v;
      if (Chp) Chp[(long)row * ldc + col] = (bf16)v;
    }
  }
}

// ------------------------------- launcher ----------------------------------

static dim3 gemm_grid(int M, int N, int Z) {
  return dim3((unsigned)((N + BN - 1) / BN), (unsigned)((M + BM - 1) / BM), (unsigned)Z);
}

extern "C" void kernel_launch(void* const* d_in, const int* in_sizes, int n_in,
                              void* d_out, int out_size, void* d_ws, size_t ws_size,
                              hipStream_t stream) {
  const int*   idx  = (const int*)d_in[0];
  const float* tok  = (const float*)d_in[1];
  const float* pos  = (const float*)d_in[2];
  const float* Wq   = (const float*)d_in[3];
  const float* Wk   = (const float*)d_in[4];
  const float* Wv   = (const float*)d_in[5];
  const float* Wo   = (const float*)d_in[6];
  const float* bo   = (const float*)d_in[7];
  const float* ln1g = (const float*)d_in[8];
  const float* ln1b = (const float*)d_in[9];
  const float* ln2g = (const float*)d_in[10];
  const float* ln2b = (const float*)d_in[11];
  const float* W1   = (const float*)d_in[12];
  const float* b1   = (const float*)d_in[13];
  const float* W2   = (const float*)d_in[14];
  const float* b2   = (const float*)d_in[15];
  const float* lnfg = (const float*)d_in[16];
  const float* lnfb = (const float*)d_in[17];
  const float* Wh   = (const float*)d_in[18];
  const float* bh   = (const float*)d_in[19];
  (void)in_sizes; (void)n_in; (void)out_size; (void)ws_size;

  // bump allocator over workspace (256B aligned)
  char* wp = (char*)d_ws;
  auto take = [&](size_t bytes) -> char* {
    char* p = wp;
    wp += (bytes + 255) & ~(size_t)255;
    return p;
  };
  bf16*  wqkv = (bf16*)take((size_t)NLAYER * E_DIM * 3 * E_DIM * 2);
  bf16*  wo   = (bf16*)take((size_t)NLAYER * E_DIM * E_DIM * 2);
  bf16*  w1   = (bf16*)take((size_t)NLAYER * E_DIM * FF * 2);
  bf16*  w2   = (bf16*)take((size_t)NLAYER * FF * E_DIM * 2);
  bf16*  wh   = (bf16*)take((size_t)E_DIM * VOCAB_PAD * 2);
  float* x    = (float*)take((size_t)NTOK * E_DIM * 4);
  bf16*  xn   = (bf16*)take((size_t)NTOK * E_DIM * 2);
  bf16*  qkv  = (bf16*)take((size_t)NTOK * 3 * E_DIM * 2);
  float* sc   = (float*)take((size_t)BSZ * NHEAD * TSEQ * TSEQ * 4);
  bf16*  pr   = (bf16*)take((size_t)BSZ * NHEAD * TSEQ * TSEQ * 2);
  bf16*  ob   = (bf16*)take((size_t)NTOK * E_DIM * 2);
  bf16*  hb   = (bf16*)take((size_t)NTOK * FF * 2);

  // ---- weight precision conversion / packing (deterministic every call)
  {
    long n = (long)NLAYER * E_DIM * 3 * E_DIM;
    pack_qkv_kernel<<<(n + 255) / 256, 256, 0, stream>>>(Wq, Wk, Wv, wqkv);
    n = (long)NLAYER * E_DIM * E_DIM;
    cvt_kernel<<<(n + 255) / 256, 256, 0, stream>>>(Wo, wo, n);
    n = (long)NLAYER * E_DIM * FF;
    cvt_kernel<<<(n + 255) / 256, 256, 0, stream>>>(W1, w1, n);
    cvt_kernel<<<(n + 255) / 256, 256, 0, stream>>>(W2, w2, n);
    n = (long)E_DIM * VOCAB_PAD;
    cvt_pad_kernel<<<(n + 255) / 256, 256, 0, stream>>>(Wh, wh, E_DIM, VOCAB_N, VOCAB_PAD);
  }

  // ---- embedding
  {
    long n = (long)NTOK * E_DIM;
    embed_kernel<<<(n + 255) / 256, 256, 0, stream>>>(idx, tok, pos, x);
  }

  const float attn_scale = 1.0f / sqrtf((float)E_DIM);  // ref scales by E^-0.5
  const int   BH = BSZ * NHEAD;

  for (int l = 0; l < NLAYER; ++l) {
    const bf16* wqkv_l = wqkv + (long)l * E_DIM * 3 * E_DIM;
    const bf16* wo_l   = wo   + (long)l * E_DIM * E_DIM;
    const bf16* w1_l   = w1   + (long)l * E_DIM * FF;
    const bf16* w2_l   = w2   + (long)l * FF * E_DIM;

    // ln1
    layernorm_kernel<<<NTOK, 256, 0, stream>>>(x, ln1g + (long)l * E_DIM,
                                               ln1b + (long)l * E_DIM, xn);
    // qkv = xn @ Wqkv  (4096 x 2304, K=768) -> bf16
    wmma_gemm_bf16<<<gemm_grid(NTOK, 3 * E_DIM, 1), 256, 0, stream>>>(
        xn, E_DIM, 0, 0,
        wqkv_l, 3 * E_DIM, 0, 0, 0,
        (float*)0, qkv, 3 * E_DIM, 0, 0,
        (const float*)0, (const float*)0, 0, 0, 0,
        NTOK, 3 * E_DIM, E_DIM, 1.0f, 0, 1);

    // scores = (Q @ K^T) * E^-0.5  per (b,h): 1024x1024, K=64
    wmma_gemm_bf16<<<gemm_grid(TSEQ, TSEQ, BH), 256, 0, stream>>>(
        qkv, 3 * E_DIM, (long)TSEQ * 3 * E_DIM, HDIM,
        qkv + E_DIM, 3 * E_DIM, (long)TSEQ * 3 * E_DIM, HDIM, 1,
        sc, (bf16*)0, TSEQ, (long)NHEAD * TSEQ * TSEQ, (long)TSEQ * TSEQ,
        (const float*)0, (const float*)0, 0, 0, 0,
        TSEQ, TSEQ, HDIM, attn_scale, 0, NHEAD);

    // causal softmax -> bf16 probs
    softmax_kernel<<<BH * TSEQ, 256, 0, stream>>>(sc, pr);

    // O = P @ V  per (b,h): 1024x64, K=1024 -> concat-heads bf16 (b,t,h*64+d)
    wmma_gemm_bf16<<<gemm_grid(TSEQ, HDIM, BH), 256, 0, stream>>>(
        pr, TSEQ, (long)NHEAD * TSEQ * TSEQ, (long)TSEQ * TSEQ,
        qkv + 2 * E_DIM, 3 * E_DIM, (long)TSEQ * 3 * E_DIM, HDIM, 0,
        (float*)0, ob, E_DIM, (long)TSEQ * E_DIM, HDIM,
        (const float*)0, (const float*)0, 0, 0, 0,
        TSEQ, HDIM, TSEQ, 1.0f, 0, NHEAD);

    // x = x + O @ Wo + bo
    wmma_gemm_bf16<<<gemm_grid(NTOK, E_DIM, 1), 256, 0, stream>>>(
        ob, E_DIM, 0, 0,
        wo_l, E_DIM, 0, 0, 0,
        x, (bf16*)0, E_DIM, 0, 0,
        bo + (long)l * E_DIM,
        x, E_DIM, 0, 0,
        NTOK, E_DIM, E_DIM, 1.0f, 0, 1);

    // ln2
    layernorm_kernel<<<NTOK, 256, 0, stream>>>(x, ln2g + (long)l * E_DIM,
                                               ln2b + (long)l * E_DIM, xn);
    // h = relu(xn @ W1 + b1)  (4096 x 3072) -> bf16
    wmma_gemm_bf16<<<gemm_grid(NTOK, FF, 1), 256, 0, stream>>>(
        xn, E_DIM, 0, 0,
        w1_l, FF, 0, 0, 0,
        (float*)0, hb, FF, 0, 0,
        b1 + (long)l * FF,
        (const float*)0, 0, 0, 0,
        NTOK, FF, E_DIM, 1.0f, 1, 1);

    // x = x + h @ W2 + b2
    wmma_gemm_bf16<<<gemm_grid(NTOK, E_DIM, 1), 256, 0, stream>>>(
        hb, FF, 0, 0,
        w2_l, E_DIM, 0, 0, 0,
        x, (bf16*)0, E_DIM, 0, 0,
        b2 + (long)l * E_DIM,
        x, E_DIM, 0, 0,
        NTOK, E_DIM, FF, 1.0f, 0, 1);
  }

  // final LN + head: logits = lnf(x) @ Wh + bh  (4096 x 50257, K=768)
  layernorm_kernel<<<NTOK, 256, 0, stream>>>(x, lnfg, lnfb, xn);
  wmma_gemm_bf16<<<gemm_grid(NTOK, VOCAB_N, 1), 256, 0, stream>>>(
      xn, E_DIM, 0, 0,
      wh, VOCAB_PAD, 0, 0, 0,
      (float*)d_out, (bf16*)0, VOCAB_N, 0, 0,
      bh,
      (const float*)0, 0, 0, 0,
      NTOK, VOCAB_N, E_DIM, 1.0f, 0, 1);
}
